// Block_78022375899354
// MI455X (gfx1250) — matrix operand order repaired and checked by reading, not compiled, not run on hardware
//
#include <hip/hip_runtime.h>

// ---------------------------------------------------------------------------
// CDNA5 (gfx1250) BF16 WMMA decoder block: causal self-attn, cross-attn,
// MLP+GELU, adapter.  Wave32.  Matrix core: V_WMMA_F32_16X16X32_BF16.
// Data movement: double-buffered LDS pipeline fed by
//   - GLOBAL_LOAD_ASYNC_TO_LDS_B128 (ASYNCcnt) for A tiles
//   - Tensor Data Mover tensor_load_to_lds (TENSORcnt) for B tiles
// with scalar staging fallback for edge tiles. DMA for tile s+1 overlaps the
// WMMA burst of tile s; one workgroup barrier per K-step.
// ---------------------------------------------------------------------------

typedef __attribute__((ext_vector_type(16))) __bf16        v16bf;
typedef __attribute__((ext_vector_type(8)))  float         v8f;
typedef __attribute__((ext_vector_type(4)))  unsigned int  ui4;
typedef __attribute__((ext_vector_type(8)))  int           i8v;
typedef __attribute__((ext_vector_type(4)))  int           i4v;

#ifndef __has_builtin
#define __has_builtin(x) 0
#endif
#if __has_builtin(__builtin_amdgcn_tensor_load_to_lds)
#define HAVE_TDM 1
#else
#define HAVE_TDM 0
#endif

union Frag16 { v16bf v; unsigned int u[8]; };

__device__ __forceinline__ unsigned short f32_to_bf16(float f) {
  unsigned int u = __float_as_uint(f);
  u += 0x7fffu + ((u >> 16) & 1u);          // round-to-nearest-even
  return (unsigned short)(u >> 16);
}

__device__ __forceinline__ float gelu_tanh(float x) {
  float x3 = x * x * x;
  return 0.5f * x * (1.0f + tanhf(0.7978845608028654f * (x + 0.044715f * x3)));
}

// ---------------------------------------------------------------------------
// Tiled BF16 GEMM with B in transposed (row-contiguous) layout:
//   D[M,N] = alpha * A[M,K] x B^T  where Bt element (n,k) = Bt[n*ldb + k].
// Epilogue: +bias[n], optional GELU, optional fp32 residual, fp32/bf16 out.
// Batched over blockIdx.z with element strides sAz/sBz/sCz/sRz.
// Block: 256 thr = 8 waves; tile 128x128; wave = 32x64 (2x4 WMMA accs).
// LDS tiles use an 80-byte row stride: 16B-aligned chunks for async/TDM DMA
// and a conflict-friendly bank walk; fragments load as ds_load_b128.
// ---------------------------------------------------------------------------
#define TM 128
#define TN 128
#define TKK 32
#define LDSS 40          /* ushorts per LDS tile row (80 bytes) */

template<bool DOGELU, bool OUTBF, bool RES>
__global__ __launch_bounds__(256)
void gemm_bt_bf16_kernel(const unsigned short* __restrict__ A, int lda, long long sAz,
                         const unsigned short* __restrict__ Bt, int ldb, long long sBz,
                         const float* __restrict__ bias,
                         float* __restrict__ Cf, unsigned short* __restrict__ Cb,
                         int ldc, long long sCz,
                         const float* __restrict__ Res, int ldr, long long sRz,
                         int M, int N, int K, float alpha)
{
  __shared__ unsigned short AsmB[2][TM * LDSS];
  __shared__ unsigned short BsmB[2][TN * LDSS];

  const int tid  = threadIdx.x;
  const int lane = tid & 31;
  const int wave = tid >> 5;
  const int wr   = wave >> 1;         // 0..3 : 32-row band
  const int wc   = wave & 1;          // 0..1 : 64-col band
  const int bm   = blockIdx.y * TM;
  const int bn   = blockIdx.x * TN;
  const long long z = blockIdx.z;

  A  += z * sAz;
  Bt += z * sBz;

  const int half = lane >> 4;
  const int l16  = lane & 15;

  // DMA eligibility: tile fully in-bounds and rows 16B-aligned.
  const bool aFull = (bm + TM <= M) && ((lda & 7) == 0);
  const bool bFull = (bn + TN <= N) && ((ldb & 7) == 0);

  // ---- stage A tile (128 x 32 bf16) into buffer `buf` ----
  auto stageA = [&](int k0, int buf) {
    unsigned short* dst = &AsmB[buf][0];
    if (aFull && (k0 + TKK <= K)) {
      // 512 chunks of 16B, 2 per lane: per-lane async DMA global->LDS.
      unsigned ldsBase = (unsigned)(unsigned long long)(const void*)dst;
      #pragma unroll
      for (int ch = tid; ch < (TM * TKK) / 8; ch += 256) {
        int r = ch >> 2, c4 = ch & 3;
        unsigned lds  = ldsBase + (unsigned)(r * (LDSS * 2) + c4 * 16);
        unsigned voff = (unsigned)(((bm + r) * lda + k0 + c4 * 8) * 2);
        asm volatile("global_load_async_to_lds_b128 %0, %1, %2"
                     :: "v"(lds), "v"(voff), "s"(A) : "memory");
      }
    } else {
      for (int e = tid; e < TM * TKK; e += 256) {
        int r = e >> 5, c = e & 31;
        int gm = bm + r, gk = k0 + c;
        unsigned short v = 0;
        if (gm < M && gk < K) v = A[(long long)gm * lda + gk];
        dst[r * LDSS + c] = v;
      }
    }
  };

  // ---- stage B tile (128 x 32 bf16, rows = n) into buffer `buf` ----
  auto stageB = [&](int k0, int buf) {
    unsigned short* dst = &BsmB[buf][0];
#if HAVE_TDM
    if (bFull && (k0 + TKK <= K)) {
      if (wave == 0) {
        unsigned ldsBase = (unsigned)(unsigned long long)(const void*)dst;
        unsigned long long ga =
            (unsigned long long)(const void*)(Bt + (long long)bn * ldb + k0);
        ui4 g0;
        g0.x = 1u;                                   // count=1, user mode
        g0.y = ldsBase;                              // LDS byte address
        g0.z = (unsigned)(ga & 0xffffffffu);         // global addr [31:0]
        g0.w = (unsigned)((ga >> 32) & 0x1ffffffu) | 0x80000000u; // [56:32] | type=2
        unsigned td0 = (unsigned)(K - k0);           // remaining extent dim0
        unsigned td1 = (unsigned)(N - bn);           // remaining extent dim1
        unsigned long long s0 = (unsigned long long)ldb;
        i8v g1;
        // data_size=1 (2B) | pad_enable | pad_interval=16DW | pad_amount=4DW
        g1[0] = (int)(0x00010000u | (1u << 20) | (3u << 22) | (3u << 25));
        g1[1] = (int)((td0 & 0xffffu) << 16);
        g1[2] = (int)((td0 >> 16) | ((td1 & 0xffffu) << 16));
        g1[3] = (int)((td1 >> 16) | ((unsigned)TKK << 16)); // tile_dim0 = 32
        g1[4] = (int)((unsigned)TN);                        // tile_dim1 = 128
        g1[5] = (int)(s0 & 0xffffffffu);
        g1[6] = (int)((s0 >> 32) & 0xffffu);
        g1[7] = 0;
        i4v zero4 = {0, 0, 0, 0};
        i8v zero8 = {0, 0, 0, 0, 0, 0, 0, 0};
        __builtin_amdgcn_tensor_load_to_lds(g0, g1, zero4, zero4, zero8, 0);
      }
    } else
#endif
    {
      for (int e = tid; e < TN * TKK; e += 256) {
        int n = e >> 5, c = e & 31;
        int gn = bn + n, gk = k0 + c;
        unsigned short v = 0;
        if (gn < N && gk < K) v = Bt[(long long)gn * ldb + gk];
        dst[n * LDSS + c] = v;
      }
    }
  };

  v8f acc[2][4];
  #pragma unroll
  for (int i = 0; i < 2; ++i)
    #pragma unroll
    for (int j = 0; j < 4; ++j)
      #pragma unroll
      for (int e = 0; e < 8; ++e) acc[i][j][e] = 0.0f;

  // ---- software pipeline: DMA(s+1) overlaps WMMA(s) ----
  stageA(0, 0);
  stageB(0, 0);
  const int nsteps = (K + TKK - 1) / TKK;

  for (int s = 0; s < nsteps; ++s) {
    const int cur = s & 1;

    // tile s landed?  (no-ops when counters are already 0)
    asm volatile("s_wait_asynccnt 0x0" ::: "memory");
#if HAVE_TDM
    if (wave == 0) {
#if __has_builtin(__builtin_amdgcn_s_wait_tensorcnt)
      __builtin_amdgcn_s_wait_tensorcnt(0);
#else
      asm volatile("s_wait_tensorcnt 0x0" ::: "memory");
#endif
    }
#endif
    __syncthreads();   // publish tile s; fence prior readers of buffer cur^1

    // kick off DMA for tile s+1 while we compute on tile s
    if (s + 1 < nsteps) {
      stageA((s + 1) * TKK, cur ^ 1);
      stageB((s + 1) * TKK, cur ^ 1);
    }

    const unsigned short* As = &AsmB[cur][0];
    const unsigned short* Bs = &BsmB[cur][0];

    // ---- fragments per ISA 16-bit layouts ----
    Frag16 bfrag[4];
    #pragma unroll
    for (int j = 0; j < 4; ++j) {
      int n = wc * 64 + j * 16 + l16;
      const unsigned int* bp = (const unsigned int*)(Bs + n * LDSS);
      #pragma unroll
      for (int i = 0; i < 8; ++i) {
        int k = ((i < 4) ? (2 * i) : (16 + 2 * (i - 4))) + 8 * half;  // B layout (same k order as A)
        bfrag[j].u[i] = bp[k >> 1];
      }
    }
    Frag16 afrag[2];
    #pragma unroll
    for (int i2 = 0; i2 < 2; ++i2) {
      int m = wr * 32 + i2 * 16 + l16;
      const unsigned int* ap = (const unsigned int*)(As + m * LDSS);
      #pragma unroll
      for (int i = 0; i < 8; ++i) {
        int k = ((i < 4) ? (2 * i) : (16 + 2 * (i - 4))) + 8 * half;  // A layout
        afrag[i2].u[i] = ap[k >> 1];
      }
    }

    #pragma unroll
    for (int i2 = 0; i2 < 2; ++i2)
      #pragma unroll
      for (int j = 0; j < 4; ++j)
        acc[i2][j] = __builtin_amdgcn_wmma_f32_16x16x32_bf16(
            false, afrag[i2].v, false, bfrag[j].v,
            (short)0, acc[i2][j], false, false);
  }

  // ---------------- epilogue (C layout: row = vgpr + 8*half, col = l16) ------
  float*          cf = Cf ? Cf + z * sCz : nullptr;
  unsigned short* cb = Cb ? Cb + z * sCz : nullptr;
  const float*    rp = RES ? Res + z * sRz : nullptr;

  #pragma unroll
  for (int i2 = 0; i2 < 2; ++i2) {
    #pragma unroll
    for (int j = 0; j < 4; ++j) {
      #pragma unroll
      for (int r = 0; r < 8; ++r) {
        int gm = bm + wr * 32 + i2 * 16 + r + 8 * half;
        int gn = bn + wc * 64 + j * 16 + l16;
        if (gm < M && gn < N) {
          float v = acc[i2][j][r] * alpha;
          if (bias) v += bias[gn];
          if (DOGELU) v = gelu_tanh(v);
          if (RES) v += rp[(long long)gm * ldr + gn];
          if (OUTBF) cb[(long long)gm * ldc + gn] = f32_to_bf16(v);
          else       cf[(long long)gm * ldc + gn] = v;
        }
      }
    }
  }
}

// ---------------------------------------------------------------------------
// LayerNorm over last dim, fp32 in -> bf16 out. One block per row.
// ---------------------------------------------------------------------------
__global__ __launch_bounds__(256)
void layernorm_bf16_kernel(const float* __restrict__ X, const float* __restrict__ g,
                           const float* __restrict__ bb, unsigned short* __restrict__ Y,
                           int Cc)
{
  long long base = (long long)blockIdx.x * Cc;
  const float* xr = X + base;
  __shared__ float red[256];
  __shared__ float smean, sinv;

  float s = 0.0f;
  for (int c = threadIdx.x; c < Cc; c += 256) s += xr[c];
  red[threadIdx.x] = s; __syncthreads();
  for (int o = 128; o > 0; o >>= 1) { if (threadIdx.x < o) red[threadIdx.x] += red[threadIdx.x + o]; __syncthreads(); }
  if (threadIdx.x == 0) smean = red[0] / (float)Cc;
  __syncthreads();
  float mean = smean;

  float vs = 0.0f;
  for (int c = threadIdx.x; c < Cc; c += 256) { float d = xr[c] - mean; vs += d * d; }
  red[threadIdx.x] = vs; __syncthreads();
  for (int o = 128; o > 0; o >>= 1) { if (threadIdx.x < o) red[threadIdx.x] += red[threadIdx.x + o]; __syncthreads(); }
  if (threadIdx.x == 0) sinv = rsqrtf(red[0] / (float)Cc + 1e-5f);
  __syncthreads();
  float inv = sinv;

  for (int c = threadIdx.x; c < Cc; c += 256)
    Y[base + c] = f32_to_bf16((xr[c] - mean) * inv * g[c] + bb[c]);
}

// ---------------------------------------------------------------------------
// Row softmax (optionally causal), fp32 -> bf16. grid = (T rows, H heads).
// ---------------------------------------------------------------------------
template<bool CAUSAL>
__global__ __launch_bounds__(256)
void softmax_bf16_kernel(const float* __restrict__ S, unsigned short* __restrict__ P, int L)
{
  const long long hs = (long long)gridDim.x * L;
  const long long base = (long long)blockIdx.y * hs + (long long)blockIdx.x * L;
  const float* s = S + base;
  unsigned short* p = P + base;
  const int valid = CAUSAL ? ((int)blockIdx.x + 1) : L;

  __shared__ float red[256];
  __shared__ float srm, srs;

  float mx = -3.0e38f;
  for (int c = threadIdx.x; c < valid; c += 256) mx = fmaxf(mx, s[c]);
  red[threadIdx.x] = mx; __syncthreads();
  for (int o = 128; o > 0; o >>= 1) { if (threadIdx.x < o) red[threadIdx.x] = fmaxf(red[threadIdx.x], red[threadIdx.x + o]); __syncthreads(); }
  if (threadIdx.x == 0) srm = red[0];
  __syncthreads();
  float rm = srm;

  float sum = 0.0f;
  for (int c = threadIdx.x; c < valid; c += 256) sum += __expf(s[c] - rm);
  red[threadIdx.x] = sum; __syncthreads();
  for (int o = 128; o > 0; o >>= 1) { if (threadIdx.x < o) red[threadIdx.x] += red[threadIdx.x + o]; __syncthreads(); }
  if (threadIdx.x == 0) srs = red[0];
  __syncthreads();
  float inv = 1.0f / srs;

  for (int c = threadIdx.x; c < L; c += 256) {
    float v = (c < valid) ? __expf(s[c] - rm) * inv : 0.0f;
    p[c] = f32_to_bf16(v);
  }
}

// ---------------------------------------------------------------------------
// Elementwise / transpose converts
// ---------------------------------------------------------------------------
__global__ void cvt_f32_bf16_kernel(const float* __restrict__ in,
                                    unsigned short* __restrict__ out, long long n)
{
  long long i = (long long)blockIdx.x * blockDim.x + threadIdx.x;
  long long st = (long long)gridDim.x * blockDim.x;
  for (; i < n; i += st) out[i] = f32_to_bf16(in[i]);
}

// out[n][k] = bf16(in[k][n]) : in is R x NC fp32, out is NC x R bf16.
__global__ __launch_bounds__(256)
void cvt_t_bf16_kernel(const float* __restrict__ in, int ldin, long long sInz,
                       unsigned short* __restrict__ out, int ldout, long long sOutz,
                       int R, int NC)
{
  __shared__ float tile[32][33];
  in  += (long long)blockIdx.z * sInz;
  out += (long long)blockIdx.z * sOutz;
  int c0 = blockIdx.x * 32;
  int r0 = blockIdx.y * 32;
  int tx = threadIdx.x & 31, ty = threadIdx.x >> 5;
  for (int rr = ty; rr < 32; rr += 8) {
    int r = r0 + rr, c = c0 + tx;
    tile[rr][tx] = (r < R && c < NC) ? in[(long long)r * ldin + c] : 0.0f;
  }
  __syncthreads();
  for (int rr = ty; rr < 32; rr += 8) {
    int n = c0 + rr, k = r0 + tx;
    if (n < NC && k < R) out[(long long)n * ldout + k] = f32_to_bf16(tile[tx][rr]);
  }
}

__global__ void add_f32_kernel(const float* __restrict__ a, const float* __restrict__ b,
                               float* __restrict__ o, long long n)
{
  long long i = (long long)blockIdx.x * blockDim.x + threadIdx.x;
  long long st = (long long)gridDim.x * blockDim.x;
  for (; i < n; i += st) o[i] = a[i] + b[i];
}

// ---------------------------------------------------------------------------
// Host launcher
// ---------------------------------------------------------------------------
static inline char* bump(char*& p, size_t bytes) {
  char* r = p;
  p += (bytes + 255) & ~(size_t)255;
  return r;
}

extern "C" void kernel_launch(void* const* d_in, const int* in_sizes, int n_in,
                              void* d_out, int out_size, void* d_ws, size_t ws_size,
                              hipStream_t stream)
{
  (void)in_sizes; (void)n_in; (void)out_size; (void)ws_size;

  const int Bb = 4, T = 1024, Te = 257, C = 1024, D = 64;
  const int C3 = 3 * C;           // 3072
  const int F  = 4 * C;           // 4096
  const int Cd = C / 4;           // 256
  const long long MT  = (long long)Bb * T;    // 4096 decoder rows
  const long long MTe = (long long)Bb * Te;   // 1028 encoder rows

  const float* x        = (const float*)d_in[0];
  const float* enc      = (const float*)d_in[1];
  const float* ln1_g    = (const float*)d_in[3];
  const float* ln1_b    = (const float*)d_in[4];
  const float* ln2_g    = (const float*)d_in[5];
  const float* ln2_b    = (const float*)d_in[6];
  const float* ln3_g    = (const float*)d_in[7];
  const float* ln3_b    = (const float*)d_in[8];
  const float* attn_w   = (const float*)d_in[9];
  const float* attn_b   = (const float*)d_in[10];
  const float* aproj_w  = (const float*)d_in[11];
  const float* aproj_b  = (const float*)d_in[12];
  const float* ca_w     = (const float*)d_in[13];
  const float* ca_b     = (const float*)d_in[14];
  const float* caproj_w = (const float*)d_in[15];
  const float* caproj_b = (const float*)d_in[16];
  const float* fc_w     = (const float*)d_in[17];
  const float* fc_b     = (const float*)d_in[18];
  const float* mproj_w  = (const float*)d_in[19];
  const float* mproj_b  = (const float*)d_in[20];
  const float* down_w   = (const float*)d_in[21];
  const float* down_b   = (const float*)d_in[22];
  const float* up_w     = (const float*)d_in[23];
  const float* up_b     = (const float*)d_in[24];

  // ---- workspace (deterministic bump allocation); weights stored TRANSPOSED
  char* p = (char*)d_ws;
  unsigned short* attnwT  = (unsigned short*)bump(p, (size_t)C * C3 * 2);   // [3C][C]
  unsigned short* aprojwT = (unsigned short*)bump(p, (size_t)C * C  * 2);   // [C][C]
  unsigned short* cawT    = (unsigned short*)bump(p, (size_t)C * C3 * 2);   // [3C][C]
  unsigned short* caprojwT= (unsigned short*)bump(p, (size_t)C * C  * 2);
  unsigned short* fcwT    = (unsigned short*)bump(p, (size_t)C * F  * 2);   // [F][C]
  unsigned short* mprojwT = (unsigned short*)bump(p, (size_t)F * C  * 2);   // [C][F]
  unsigned short* downwT  = (unsigned short*)bump(p, (size_t)C * Cd * 2);   // [Cd][C]
  unsigned short* upwT    = (unsigned short*)bump(p, (size_t)Cd * C * 2);   // [C][Cd]

  float*          xcur    = (float*)bump(p, (size_t)MT * C * 4);
  unsigned short* xln_bf  = (unsigned short*)bump(p, (size_t)MT * C * 2);
  float*          qkv_f   = (float*)bump(p, (size_t)MT * C3 * 4);
  unsigned short* qkv_bf  = (unsigned short*)bump(p, (size_t)MT * C3 * 2);
  float*          scores  = (float*)bump(p, (size_t)16 * T * T * 4);  // reused by cross
  unsigned short* probs   = (unsigned short*)bump(p, (size_t)16 * T * T * 2);
  unsigned short* vt      = (unsigned short*)bump(p, (size_t)Bb * 16 * D * T * 2);   // [b][h][64][T]
  unsigned short* vte     = (unsigned short*)bump(p, (size_t)Bb * 16 * D * Te * 2);  // [b][h][64][Te]
  float*          attnout = (float*)bump(p, (size_t)MT * C * 4);
  unsigned short* attnout_bf = (unsigned short*)bump(p, (size_t)MT * C * 2);
  float*          qdec_f  = (float*)bump(p, (size_t)MT * C * 4);
  unsigned short* qdec_bf = (unsigned short*)bump(p, (size_t)MT * C * 2);
  unsigned short* enc_bf  = (unsigned short*)bump(p, (size_t)MTe * C * 2);
  float*          kvenc_f = (float*)bump(p, (size_t)MTe * 2 * C * 4);
  unsigned short* kvenc_bf= (unsigned short*)bump(p, (size_t)MTe * 2 * C * 2);
  float*          caout   = (float*)bump(p, (size_t)MT * C * 4);
  unsigned short* caout_bf= (unsigned short*)bump(p, (size_t)MT * C * 2);
  unsigned short* g1_bf   = (unsigned short*)bump(p, (size_t)MT * F * 2);
  float*          hbuf    = (float*)bump(p, (size_t)MT * C * 4);
  unsigned short* h_bf    = (unsigned short*)bump(p, (size_t)MT * C * 2);
  unsigned short* g2_bf   = (unsigned short*)bump(p, (size_t)MT * Cd * 2);
  float*          adapt   = (float*)bump(p, (size_t)MT * C * 4);

  auto cvt = [&](const float* src, unsigned short* dst, long long n) {
    int blocks = (int)((n + 255) / 256); if (blocks > 16384) blocks = 16384;
    cvt_f32_bf16_kernel<<<blocks, 256, 0, stream>>>(src, dst, n);
  };

  // ---- weights: fp32 [K][N] -> bf16 transposed [N][K] ----
  cvt_t_bf16_kernel<<<dim3(C3/32, C/32, 1), 256, 0, stream>>>(attn_w,   C3, 0, attnwT,   C, 0, C, C3);
  cvt_t_bf16_kernel<<<dim3(C/32,  C/32, 1), 256, 0, stream>>>(aproj_w,  C,  0, aprojwT,  C, 0, C, C);
  cvt_t_bf16_kernel<<<dim3(C3/32, C/32, 1), 256, 0, stream>>>(ca_w,     C3, 0, cawT,     C, 0, C, C3);
  cvt_t_bf16_kernel<<<dim3(C/32,  C/32, 1), 256, 0, stream>>>(caproj_w, C,  0, caprojwT, C, 0, C, C);
  cvt_t_bf16_kernel<<<dim3(F/32,  C/32, 1), 256, 0, stream>>>(fc_w,     F,  0, fcwT,     C, 0, C, F);
  cvt_t_bf16_kernel<<<dim3(C/32,  F/32, 1), 256, 0, stream>>>(mproj_w,  C,  0, mprojwT,  F, 0, F, C);
  cvt_t_bf16_kernel<<<dim3(Cd/32, C/32, 1), 256, 0, stream>>>(down_w,   Cd, 0, downwT,   C, 0, C, Cd);
  cvt_t_bf16_kernel<<<dim3(C/32,  Cd/32,1), 256, 0, stream>>>(up_w,     C,  0, upwT,     Cd,0, Cd, C);
  cvt(enc, enc_bf, (long long)MTe * C);

  // ---- residual stream init ----
  (void)hipMemcpyAsync(xcur, x, (size_t)MT * C * 4, hipMemcpyDeviceToDevice, stream);

  // ================= self-attention =================
  layernorm_bf16_kernel<<<(int)MT, 256, 0, stream>>>(xcur, ln1_g, ln1_b, xln_bf, C);

  gemm_bt_bf16_kernel<false, false, false><<<dim3(C3/128, (int)MT/128, 1), 256, 0, stream>>>(
      xln_bf, C, 0, attnwT, C, 0, attn_b,
      qkv_f, nullptr, C3, 0, nullptr, 0, 0, (int)MT, C3, C, 1.0f);
  cvt(qkv_f, qkv_bf, (long long)MT * C3);

  for (int b = 0; b < Bb; ++b) {
    // V^T copies: [h][64][T] from qkv_f
    cvt_t_bf16_kernel<<<dim3(2, T/32, 16), 256, 0, stream>>>(
        qkv_f + (long long)b * T * C3 + 2 * C, C3, 64,
        vt + (long long)b * 16 * D * T, T, (long long)D * T, T, D);

    const unsigned short* Qb = qkv_bf + (long long)b * T * C3;
    const unsigned short* Kb = Qb + C;
    // S = Q K^T / sqrt(D)   (async-A + TDM-B eligible everywhere)
    gemm_bt_bf16_kernel<false, false, false><<<dim3(8, 8, 16), 256, 0, stream>>>(
        Qb, C3, 64, Kb, C3, 64, nullptr,
        scores, nullptr, T, (long long)T * T, nullptr, 0, 0, T, T, D, 0.125f);
    softmax_bf16_kernel<true><<<dim3(T, 16), 256, 0, stream>>>(scores, probs, T);
    // O = P V  (merged-head output)
    gemm_bt_bf16_kernel<false, false, false><<<dim3(1, 8, 16), 256, 0, stream>>>(
        probs, T, (long long)T * T, vt + (long long)b * 16 * D * T, T, (long long)D * T, nullptr,
        attnout + (long long)b * T * C, nullptr, C, 64, nullptr, 0, 0, T, D, T, 1.0f);
  }
  cvt(attnout, attnout_bf, (long long)MT * C);
  gemm_bt_bf16_kernel<false, false, true><<<dim3(C/128, (int)MT/128, 1), 256, 0, stream>>>(
      attnout_bf, C, 0, aprojwT, C, 0, aproj_b,
      xcur, nullptr, C, 0, xcur, C, 0, (int)MT, C, C, 1.0f);

  // ================= cross-attention =================
  layernorm_bf16_kernel<<<(int)MT, 256, 0, stream>>>(xcur, ln2_g, ln2_b, xln_bf, C);

  gemm_bt_bf16_kernel<false, false, false><<<dim3(C/128, (int)MT/128, 1), 256, 0, stream>>>(
      xln_bf, C, 0, cawT, C, 0, ca_b,
      qdec_f, nullptr, C, 0, nullptr, 0, 0, (int)MT, C, C, 1.0f);
  cvt(qdec_f, qdec_bf, (long long)MT * C);

  gemm_bt_bf16_kernel<false, false, false><<<dim3(16, ((int)MTe+127)/128, 1), 256, 0, stream>>>(
      enc_bf, C, 0, cawT + (long long)C * C, C, 0, ca_b + C,
      kvenc_f, nullptr, 2 * C, 0, nullptr, 0, 0, (int)MTe, 2 * C, C, 1.0f);
  cvt(kvenc_f, kvenc_bf, (long long)MTe * 2 * C);

  for (int b = 0; b < Bb; ++b) {
    cvt_t_bf16_kernel<<<dim3(2, (Te+31)/32, 16), 256, 0, stream>>>(
        kvenc_f + (long long)b * Te * 2 * C + C, 2 * C, 64,
        vte + (long long)b * 16 * D * Te, Te, (long long)D * Te, Te, D);

    const unsigned short* Qd = qdec_bf + (long long)b * T * C;
    const unsigned short* Ke = kvenc_bf + (long long)b * Te * (2 * C);
    gemm_bt_bf16_kernel<false, false, false><<<dim3((Te+127)/128, 8, 16), 256, 0, stream>>>(
        Qd, C, 64, Ke, 2 * C, 64, nullptr,
        scores, nullptr, Te, (long long)T * Te, nullptr, 0, 0, T, Te, D, 0.125f);
    softmax_bf16_kernel<false><<<dim3(T, 16), 256, 0, stream>>>(scores, probs, Te);
    gemm_bt_bf16_kernel<false, false, false><<<dim3(1, 8, 16), 256, 0, stream>>>(
        probs, Te, (long long)T * Te, vte + (long long)b * 16 * D * Te, Te, (long long)D * Te, nullptr,
        caout + (long long)b * T * C, nullptr, C, 64, nullptr, 0, 0, T, D, Te, 1.0f);
  }
  cvt(caout, caout_bf, (long long)MT * C);
  gemm_bt_bf16_kernel<false, false, true><<<dim3(C/128, (int)MT/128, 1), 256, 0, stream>>>(
      caout_bf, C, 0, caprojwT, C, 0, caproj_b,
      xcur, nullptr, C, 0, xcur, C, 0, (int)MT, C, C, 1.0f);

  // ================= MLP + adapter =================
  layernorm_bf16_kernel<<<(int)MT, 256, 0, stream>>>(xcur, ln3_g, ln3_b, xln_bf, C);

  gemm_bt_bf16_kernel<true, true, false><<<dim3(F/128, (int)MT/128, 1), 256, 0, stream>>>(
      xln_bf, C, 0, fcwT, C, 0, fc_b,
      nullptr, g1_bf, F, 0, nullptr, 0, 0, (int)MT, F, C, 1.0f);
  gemm_bt_bf16_kernel<false, false, false><<<dim3(C/128, (int)MT/128, 1), 256, 0, stream>>>(
      g1_bf, F, 0, mprojwT, F, 0, mproj_b,
      hbuf, nullptr, C, 0, nullptr, 0, 0, (int)MT, C, F, 1.0f);
  cvt(hbuf, h_bf, (long long)MT * C);
  gemm_bt_bf16_kernel<true, true, false><<<dim3(Cd/128, (int)MT/128, 1), 256, 0, stream>>>(
      h_bf, C, 0, downwT, C, 0, down_b,
      nullptr, g2_bf, Cd, 0, nullptr, 0, 0, (int)MT, Cd, C, 1.0f);
  gemm_bt_bf16_kernel<false, false, true><<<dim3(C/128, (int)MT/128, 1), 256, 0, stream>>>(
      g2_bf, Cd, 0, upwT, Cd, 0, up_b,
      adapt, nullptr, C, 0, hbuf, C, 0, (int)MT, C, Cd, 1.0f);

  {
    long long n = (long long)MT * C;
    int blocks = (int)((n + 255) / 256); if (blocks > 16384) blocks = 16384;
    add_f32_kernel<<<blocks, 256, 0, stream>>>(xcur, adapt, (float*)d_out, n);
  }
}
